// BFS_Neural_Execution_85925115724476
// MI455X (gfx1250) — matrix operand order repaired and checked by reading, not compiled, not run on hardware
//
#include <hip/hip_runtime.h>
#include <hip/hip_bf16.h>
#include <math.h>

#define N_NODES 50000
#define N_EDGES 600000
#define HD      128

// N_NODES/16 = 3125 node tiles, N_EDGES/16 = 37500 edge tiles (both exact).

typedef __attribute__((ext_vector_type(16))) __bf16         v16bf;
typedef __attribute__((ext_vector_type(8)))  __bf16         v8bf;
typedef __attribute__((ext_vector_type(8)))  float          v8f;
typedef __attribute__((ext_vector_type(8)))  unsigned short v8us;

// Native f32 -> bf16 (lowers to v_cvt_pk_bf16_f32 on gfx1250, RNE in HW).
__device__ __forceinline__ unsigned short f2bf(float f) {
    return __builtin_bit_cast(unsigned short, (__bf16)f);
}
__device__ __forceinline__ float bf2f(unsigned short s) {
    return __uint_as_float(((unsigned int)s) << 16);
}

// Build a 16x32 bf16 A/B fragment from two contiguous 16-byte chunks.
__device__ __forceinline__ v16bf frag_bf(const unsigned short* p0, const unsigned short* p1) {
    v8bf lo = __builtin_bit_cast(v8bf, *(const v8us*)p0);
    v8bf hi = __builtin_bit_cast(v8bf, *(const v8us*)p1);
    return __builtin_shufflevector(lo, hi, 0,1,2,3,4,5,6,7,8,9,10,11,12,13,14,15);
}
__device__ __forceinline__ v8bf cvt8(const float* p) {
    float4 a = *(const float4*)p;
    float4 b = *(const float4*)(p + 4);
    v8bf r;
    r[0]=(__bf16)a.x; r[1]=(__bf16)a.y; r[2]=(__bf16)a.z; r[3]=(__bf16)a.w;
    r[4]=(__bf16)b.x; r[5]=(__bf16)b.y; r[6]=(__bf16)b.z; r[7]=(__bf16)b.w;
    return r;
}
__device__ __forceinline__ v16bf frag_f32(const float* p0, const float* p1) {
    v8bf lo = cvt8(p0);
    v8bf hi = cvt8(p1);
    return __builtin_shufflevector(lo, hi, 0,1,2,3,4,5,6,7,8,9,10,11,12,13,14,15);
}

#define WMMA_BF16(acc, a, b) \
    __builtin_amdgcn_wmma_f32_16x16x32_bf16(false, (a), false, (b), (short)0, (acc), false, false)

// ---------------------------------------------------------------------------
// K0: convert+transpose weights to bf16 [col][k]; zero agg and hsum.
// ---------------------------------------------------------------------------
__global__ void prep_kernel(const float* __restrict__ enc_w, const float* __restrict__ M_w,
                            const float* __restrict__ U_w,
                            unsigned short* __restrict__ enc_wT, unsigned short* __restrict__ M_wT,
                            unsigned short* __restrict__ U_wT,
                            float* __restrict__ agg, float* __restrict__ hsum) {
    long long i      = (long long)blockIdx.x * blockDim.x + threadIdx.x;
    long long stride = (long long)gridDim.x * blockDim.x;
    for (long long j = i; j < (long long)N_NODES * HD; j += stride) agg[j] = 0.0f;
    if (i < HD) hsum[i] = 0.0f;
    for (long long j = i; j < 128 * 128; j += stride) {
        int c = (int)(j >> 7), k = (int)(j & 127);
        enc_wT[c * 128 + k] = f2bf(enc_w[(k + 1) * 128 + c]);   // k=0 of concat is x
    }
    for (long long j = i; j < 128 * 256; j += stride) {
        int c = (int)(j >> 8), k = (int)(j & 255);
        M_wT[c * 256 + k] = f2bf(M_w[k * 128 + c]);
        U_wT[c * 256 + k] = f2bf(U_w[k * 128 + c]);
    }
}

// ---------------------------------------------------------------------------
// K1: encoder  z = relu([x, pre_h] @ enc_w + enc_b)  -> z stored as bf16
// ---------------------------------------------------------------------------
__global__ void __launch_bounds__(256)
enc_kernel(const float* __restrict__ x, const float* __restrict__ pre_h,
           const unsigned short* __restrict__ enc_wTg, const float* __restrict__ enc_w,
           const float* __restrict__ enc_b, unsigned short* __restrict__ zb) {
    __shared__ unsigned short sW[128 * 128];                    // 32 KB
    {
        const uint4* g = (const uint4*)enc_wTg;
        uint4* s = (uint4*)sW;
        for (int i = threadIdx.x; i < 128 * 128 / 8; i += blockDim.x) s[i] = g[i];
    }
    __syncthreads();

    int wave = threadIdx.x >> 5;
    int tile = blockIdx.x * 8 + wave;
    if (tile >= N_NODES / 16) return;
    int lane = threadIdx.x & 31;
    int m16  = lane & 15, half = lane >> 4;

    const float* ph = pre_h + (long long)(tile * 16 + m16) * HD;
    __builtin_prefetch(ph, 0, 3);

    v8f acc[8];
    #pragma unroll
    for (int ct = 0; ct < 8; ct++) acc[ct] = (v8f)0.0f;

    #pragma unroll
    for (int ks = 0; ks < 4; ks++) {
        int kb = ks * 32;
        v16bf a = frag_f32(ph + kb + half * 8, ph + kb + 16 + half * 8);
        v16bf bf[8];
        #pragma unroll
        for (int ct = 0; ct < 8; ct++) {                        // batch the 16 ds_load_b128
            const unsigned short* wp = sW + (ct * 16 + m16) * 128 + kb + half * 8;
            bf[ct] = frag_bf(wp, wp + 16);
        }
        #pragma unroll
        for (int ct = 0; ct < 8; ct++)
            acc[ct] = WMMA_BF16(acc[ct], a, bf[ct]);
    }

    float xr[8];
    #pragma unroll
    for (int r = 0; r < 8; r++) xr[r] = x[tile * 16 + r + 8 * half];

    #pragma unroll
    for (int ct = 0; ct < 8; ct++) {
        int   n  = ct * 16 + m16;
        float bb = enc_b[n];
        float w0 = enc_w[n];                                    // row 0 of enc_w (x weight)
        #pragma unroll
        for (int r = 0; r < 8; r++) {
            float v = fmaxf(acc[ct][r] + bb + xr[r] * w0, 0.0f);
            zb[(long long)(tile * 16 + r + 8 * half) * HD + n] = f2bf(v);
        }
    }
}

// ---------------------------------------------------------------------------
// K2: edge MLP + scatter-max.  m = relu([z[tgt],z[src],ea] @ M_w + M_b)
//     agg[t] = max(agg[t], m)   (ReLU => nonneg => int-ordered atomic max)
// ---------------------------------------------------------------------------
__global__ void __launch_bounds__(256)
edge_kernel(const unsigned short* __restrict__ zb, const int* __restrict__ eidx,
            const float* __restrict__ edge_attr, const unsigned short* __restrict__ M_wTg,
            const float* __restrict__ M_w, const float* __restrict__ M_b,
            float* __restrict__ agg) {
    __shared__ unsigned short sW[128 * 256];                    // 64 KB
    {
        const uint4* g = (const uint4*)M_wTg;
        uint4* s = (uint4*)sW;
        for (int i = threadIdx.x; i < 128 * 256 / 8; i += blockDim.x) s[i] = g[i];
    }
    __syncthreads();

    int wave = threadIdx.x >> 5;
    int tile = blockIdx.x * 8 + wave;
    if (tile >= N_EDGES / 16) return;
    int lane = threadIdx.x & 31;
    int m16  = lane & 15, half = lane >> 4;

    const int* src = eidx;
    const int* tgt = eidx + N_EDGES;

    int eA = tile * 16 + m16;
    const unsigned short* zt = zb + (long long)tgt[eA] * HD;
    const unsigned short* zs = zb + (long long)src[eA] * HD;
    __builtin_prefetch(zt, 0, 3);                               // global_prefetch_b8: pull the two
    __builtin_prefetch(zs, 0, 3);                               // gathered z rows toward the WGP

    v8f acc[8];
    #pragma unroll
    for (int ct = 0; ct < 8; ct++) acc[ct] = (v8f)0.0f;

    #pragma unroll
    for (int ks = 0; ks < 8; ks++) {
        const unsigned short* base = (ks < 4) ? zt : zs;        // K 0..127 = z_tgt, 128..255 = z_src
        int kb = (ks & 3) * 32;
        v16bf a = frag_bf(base + kb + half * 8, base + kb + 16 + half * 8);
        v16bf bf[8];
        #pragma unroll
        for (int ct = 0; ct < 8; ct++) {                        // batch the 16 ds_load_b128
            const unsigned short* wp = sW + (ct * 16 + m16) * 256 + ks * 32 + half * 8;
            bf[ct] = frag_bf(wp, wp + 16);
        }
        #pragma unroll
        for (int ct = 0; ct < 8; ct++)
            acc[ct] = WMMA_BF16(acc[ct], a, bf[ct]);
    }

    float ea[8];
    int   tg[8];
    #pragma unroll
    for (int r = 0; r < 8; r++) {
        int e  = tile * 16 + r + 8 * half;
        ea[r]  = edge_attr[e];
        tg[r]  = tgt[e];
    }
    const float* w256 = M_w + 256 * 128;                        // edge_attr row of M_w

    #pragma unroll
    for (int ct = 0; ct < 8; ct++) {
        int   n  = ct * 16 + m16;
        float bb = M_b[n];
        float w2 = w256[n];
        #pragma unroll
        for (int r = 0; r < 8; r++) {
            float v = fmaxf(acc[ct][r] + bb + ea[r] * w2, 0.0f);
            atomicMax((int*)(agg + (long long)tg[r] * HD + n), __float_as_int(v));
        }
    }
}

// ---------------------------------------------------------------------------
// K3: update  h = relu([z, agg] @ U_w + U_b)  -> fp32 into d_out
// ---------------------------------------------------------------------------
__global__ void __launch_bounds__(256)
u_kernel(const unsigned short* __restrict__ zb, const float* __restrict__ agg,
         const unsigned short* __restrict__ U_wTg, const float* __restrict__ U_b,
         float* __restrict__ h_out) {
    __shared__ unsigned short sW[128 * 256];                    // 64 KB
    {
        const uint4* g = (const uint4*)U_wTg;
        uint4* s = (uint4*)sW;
        for (int i = threadIdx.x; i < 128 * 256 / 8; i += blockDim.x) s[i] = g[i];
    }
    __syncthreads();

    int wave = threadIdx.x >> 5;
    int tile = blockIdx.x * 8 + wave;
    if (tile >= N_NODES / 16) return;
    int lane = threadIdx.x & 31;
    int m16  = lane & 15, half = lane >> 4;

    const unsigned short* zr = zb  + (long long)(tile * 16 + m16) * HD;
    const float*          ar = agg + (long long)(tile * 16 + m16) * HD;
    __builtin_prefetch(zr, 0, 3);
    __builtin_prefetch(ar, 0, 3);

    v8f acc[8];
    #pragma unroll
    for (int ct = 0; ct < 8; ct++) acc[ct] = (v8f)0.0f;

    #pragma unroll
    for (int ks = 0; ks < 8; ks++) {
        int kb = (ks & 3) * 32;
        v16bf a = (ks < 4) ? frag_bf (zr + kb + half * 8, zr + kb + 16 + half * 8)
                           : frag_f32(ar + kb + half * 8, ar + kb + 16 + half * 8);
        v16bf bf[8];
        #pragma unroll
        for (int ct = 0; ct < 8; ct++) {                        // batch the 16 ds_load_b128
            const unsigned short* wp = sW + (ct * 16 + m16) * 256 + ks * 32 + half * 8;
            bf[ct] = frag_bf(wp, wp + 16);
        }
        #pragma unroll
        for (int ct = 0; ct < 8; ct++)
            acc[ct] = WMMA_BF16(acc[ct], a, bf[ct]);
    }

    #pragma unroll
    for (int ct = 0; ct < 8; ct++) {
        int   n  = ct * 16 + m16;
        float bb = U_b[n];
        #pragma unroll
        for (int r = 0; r < 8; r++) {
            float v = fmaxf(acc[ct][r] + bb, 0.0f);
            h_out[(long long)(tile * 16 + r + 8 * half) * HD + n] = v;
        }
    }
}

// ---------------------------------------------------------------------------
// K4: decoder  y = sigmoid([z,h] @ dec_w + dec_b) — one wave per node
// ---------------------------------------------------------------------------
__global__ void __launch_bounds__(256)
dec_kernel(const unsigned short* __restrict__ zb, const float* __restrict__ h,
           const float* __restrict__ dec_w, const float* __restrict__ dec_b,
           float* __restrict__ y) {
    int wave = threadIdx.x >> 5;
    int lane = threadIdx.x & 31;
    int node = blockIdx.x * 8 + wave;
    if (node >= N_NODES) return;
    const unsigned short* zr = zb + (long long)node * HD;
    const float*          hr = h  + (long long)node * HD;
    float t = 0.0f;
    #pragma unroll
    for (int k = lane; k < HD; k += 32) t += bf2f(zr[k]) * dec_w[k];
    #pragma unroll
    for (int k = lane; k < HD; k += 32) t += hr[k] * dec_w[128 + k];
    #pragma unroll
    for (int o = 16; o > 0; o >>= 1) t += __shfl_down(t, o, 32);
    if (lane == 0) y[node] = 1.0f / (1.0f + expf(-(t + dec_b[0])));
}

// ---------------------------------------------------------------------------
// K5: column sums of h (for h_mean)
// ---------------------------------------------------------------------------
__global__ void hsum_kernel(const float* __restrict__ h, float* __restrict__ hsum) {
    int c = threadIdx.x;                                        // 128 threads
    float s = 0.0f;
    for (int n = blockIdx.x; n < N_NODES; n += gridDim.x)
        s += h[(long long)n * HD + c];
    atomicAdd(hsum + c, s);
}

// ---------------------------------------------------------------------------
// K6: terminator.  mean([h,h_mean]@ter_w + ter_b) = h_mean·(w0+w1) + ter_b
// ---------------------------------------------------------------------------
__global__ void ter_kernel(const float* __restrict__ hsum, const float* __restrict__ ter_w,
                           const float* __restrict__ ter_b, float* __restrict__ out) {
    __shared__ float red[128];
    int c = threadIdx.x;
    float hm = hsum[c] * (1.0f / (float)N_NODES);
    red[c] = hm * (ter_w[c] + ter_w[128 + c]);
    __syncthreads();
    #pragma unroll
    for (int o = 64; o > 0; o >>= 1) {
        if (c < o) red[c] += red[c + o];
        __syncthreads();
    }
    if (c == 0) out[0] = red[0] + ter_b[0];
}

// ---------------------------------------------------------------------------
extern "C" void kernel_launch(void* const* d_in, const int* in_sizes, int n_in,
                              void* d_out, int out_size, void* d_ws, size_t ws_size,
                              hipStream_t stream) {
    const float* x         = (const float*)d_in[0];
    const float* pre_h     = (const float*)d_in[1];
    const float* edge_attr = (const float*)d_in[2];
    const float* enc_w     = (const float*)d_in[3];
    const float* enc_b     = (const float*)d_in[4];
    const float* M_w       = (const float*)d_in[5];
    const float* M_b       = (const float*)d_in[6];
    const float* U_w       = (const float*)d_in[7];
    const float* U_b       = (const float*)d_in[8];
    const float* dec_w     = (const float*)d_in[9];
    const float* dec_b     = (const float*)d_in[10];
    const float* ter_w     = (const float*)d_in[11];
    const float* ter_b     = (const float*)d_in[12];
    const int*   eidx      = (const int*)d_in[13];

    // d_out layout: h [N*H] | y [N] | ter [1]
    float* h_out = (float*)d_out;
    float* y_out = h_out + (size_t)N_NODES * HD;
    float* t_out = y_out + N_NODES;

    // workspace carve (all 16B aligned)
    char* ws = (char*)d_ws;
    size_t off = 0;
    unsigned short* zb     = (unsigned short*)(ws + off); off += (size_t)N_NODES * HD * 2;
    float*          agg    = (float*)(ws + off);          off += (size_t)N_NODES * HD * 4;
    unsigned short* enc_wT = (unsigned short*)(ws + off); off += 128 * 128 * 2;
    unsigned short* M_wT   = (unsigned short*)(ws + off); off += 128 * 256 * 2;
    unsigned short* U_wT   = (unsigned short*)(ws + off); off += 128 * 256 * 2;
    float*          hsum   = (float*)(ws + off);          off += 128 * 4;
    (void)in_sizes; (void)n_in; (void)out_size; (void)ws_size;

    prep_kernel<<<2048, 256, 0, stream>>>(enc_w, M_w, U_w, enc_wT, M_wT, U_wT, agg, hsum);
    enc_kernel<<<(N_NODES / 16 + 7) / 8, 256, 0, stream>>>(x, pre_h, enc_wT, enc_w, enc_b, zb);
    edge_kernel<<<(N_EDGES / 16 + 7) / 8, 256, 0, stream>>>(zb, eidx, edge_attr, M_wT, M_w, M_b, agg);
    u_kernel<<<(N_NODES / 16 + 7) / 8, 256, 0, stream>>>(zb, agg, U_wT, U_b, h_out);
    dec_kernel<<<(N_NODES + 7) / 8, 256, 0, stream>>>(zb, h_out, dec_w, dec_b, y_out);
    hsum_kernel<<<512, 128, 0, stream>>>(h_out, hsum);
    ter_kernel<<<1, 128, 0, stream>>>(hsum, ter_w, ter_b, t_out);
}
